// MockBitNetLayer_89489938580120
// MI455X (gfx1250) — matrix operand order repaired and checked by reading, not compiled, not run on hardware
//
#include <hip/hip_runtime.h>

// Problem shape (fixed by the reference)
#define TOKENS 8192
#define DIN    4096
#define DOUT   16384

// GEMM tiling: block 128x256, 8 waves (2x4), wave tile 64x64, K-step 32
#define BM 128
#define BN 256
#define BK 32
#define NIT (DIN / BK)                       // 128 K-iterations

#define LDS_STRIDE_H 40                      // halfwords per 32-col row (8 pad) -> conflict-free b128
#define ROW_BYTES    (LDS_STRIDE_H * 2)      // 80 B
#define ABUF_BYTES   (BM * ROW_BYTES)        // 10240 B per A buffer
#define BBUF_BYTES   (BN * ROW_BYTES)        // 20480 B per B buffer
#define SMEM_BYTES   (2 * ABUF_BYTES + 2 * BBUF_BYTES)   // 61440 B dynamic LDS

typedef __attribute__((ext_vector_type(16))) __bf16        v16bf;
typedef __attribute__((ext_vector_type(8)))  float         v8f;
typedef __attribute__((ext_vector_type(4)))  unsigned int  u32x4;
typedef __attribute__((address_space(3)))    char*         lds_ptr_t;

// ---------- CDNA5 async copy: global -> LDS, per-lane b128, tracked by ASYNCcnt ----------
static __device__ __forceinline__ void async_copy_b128(char* lds, const void* g) {
  asm volatile("global_load_async_to_lds_b128 %0, %1, off"
               :: "v"((lds_ptr_t)lds), "v"(g)
               : "memory");
}
#define WAIT_ASYNC(n) asm volatile("s_wait_asynccnt %0" :: "n"(n) : "memory")

// ---------- helpers ----------
static __device__ __forceinline__ unsigned short f32_to_bf16_rne(float f) {
  union { float f; unsigned int u; } c; c.f = f;
  unsigned int u = c.u;
  u += 0x7FFFu + ((u >> 16) & 1u);           // RNE truncation (finite inputs)
  return (unsigned short)(u >> 16);
}

static __device__ __forceinline__ v16bf load_frag(const char* lo, const char* hi) {
  union { u32x4 u[2]; v16bf v; } c;
  c.u[0] = *(const u32x4*)lo;                // ds_load_b128
  c.u[1] = *(const u32x4*)hi;                // ds_load_b128
  return c.v;
}

// ---------- kernel 1: x (f32) -> bf16 ----------
__global__ __launch_bounds__(256)
void quant_x_kernel(const float* __restrict__ X, unsigned short* __restrict__ Xb) {
  size_t i = ((size_t)blockIdx.x * 256 + threadIdx.x) * 8;
  float4 a = *(const float4*)(X + i);
  float4 b = *(const float4*)(X + i + 4);
  ushort4 pa, pb;
  pa.x = f32_to_bf16_rne(a.x); pa.y = f32_to_bf16_rne(a.y);
  pa.z = f32_to_bf16_rne(a.z); pa.w = f32_to_bf16_rne(a.w);
  pb.x = f32_to_bf16_rne(b.x); pb.y = f32_to_bf16_rne(b.y);
  pb.z = f32_to_bf16_rne(b.z); pb.w = f32_to_bf16_rne(b.w);
  *(ushort4*)(Xb + i)     = pa;
  *(ushort4*)(Xb + i + 4) = pb;
}

// ---------- kernel 2: per-row scale = mean|W|, signW -> bf16 {+1,-1,0} ----------
__global__ __launch_bounds__(256)
void quant_w_kernel(const float* __restrict__ W,
                    unsigned short* __restrict__ Wb,
                    float* __restrict__ scale) {
  __shared__ float red[256];
  const int row = blockIdx.x;
  const int tid = threadIdx.x;
  const float*    wrow = W  + (size_t)row * DIN;
  unsigned short* brow = Wb + (size_t)row * DIN;
  const int base = tid * 16;                 // 256 threads * 16 = 4096
  float s = 0.f;
#pragma unroll
  for (int j = 0; j < 16; j += 4) {
    float4 w = *(const float4*)(wrow + base + j);
    s += fabsf(w.x) + fabsf(w.y) + fabsf(w.z) + fabsf(w.w);
    ushort4 sg;
    sg.x = w.x > 0.f ? 0x3F80u : (w.x < 0.f ? 0xBF80u : 0u);
    sg.y = w.y > 0.f ? 0x3F80u : (w.y < 0.f ? 0xBF80u : 0u);
    sg.z = w.z > 0.f ? 0x3F80u : (w.z < 0.f ? 0xBF80u : 0u);
    sg.w = w.w > 0.f ? 0x3F80u : (w.w < 0.f ? 0xBF80u : 0u);
    *(ushort4*)(brow + base + j) = sg;
  }
  red[tid] = s;
  __syncthreads();
#pragma unroll
  for (int off = 128; off > 0; off >>= 1) {
    if (tid < off) red[tid] += red[tid + off];
    __syncthreads();
  }
  if (tid == 0) scale[row] = red[0] * (1.0f / (float)DIN);
}

// ---------- kernel 3: out = (Xb @ Wb^T) * scale[n] + bias[n], async double-buffered WMMA ----------
__global__ __launch_bounds__(256)
void bitnet_wmma_kernel(const unsigned short* __restrict__ Xb,
                        const unsigned short* __restrict__ Wb,
                        const float* __restrict__ scale,
                        const float* __restrict__ bias,
                        float* __restrict__ Out) {
  extern __shared__ char smem[];
  char* ldsA0 = smem;
  char* ldsA1 = smem + ABUF_BYTES;
  char* ldsB0 = smem + 2 * ABUF_BYTES;
  char* ldsB1 = smem + 2 * ABUF_BYTES + BBUF_BYTES;

  const int tid  = threadIdx.x;
  const int lane = tid & 31;
  const int wave = tid >> 5;     // 8 waves
  const int wm   = wave >> 2;    // 0..1 -> 64-row slice
  const int wn   = wave & 3;     // 0..3 -> 64-col slice
  const int m0   = blockIdx.y * BM;
  const int n0   = blockIdx.x * BN;

  // async tile mapping: 4 threads/row, one 16B chunk each; A covers 128 rows in 2
  // passes of 64, B covers 256 rows in 4 passes of 64.
  const int ldRow = tid >> 2;          // 0..63
  const int ldCol = (tid & 3) * 8;     // 0,8,16,24 (bf16 units)

  const unsigned short* gA = Xb + (size_t)(m0 + ldRow) * DIN + ldCol;
  const unsigned short* gB = Wb + (size_t)(n0 + ldRow) * DIN + ldCol;
  const unsigned ldOff = (unsigned)(ldRow * ROW_BYTES + ldCol * 2);

  v8f acc[4][4];
#pragma unroll
  for (int mt = 0; mt < 4; ++mt)
#pragma unroll
    for (int nt = 0; nt < 4; ++nt)
      acc[mt][nt] = (v8f){0.f, 0.f, 0.f, 0.f, 0.f, 0.f, 0.f, 0.f};

  // fragment addressing (ISA 16-bit wave32 layouts)
  const int frow  = lane & 15;
  const int aKoff = (lane >> 4) * 16;   // bytes: A lanes16-31 hold K=8..15 / 24..31
  const int bKoff = (lane >> 4) * 16;   // bytes: B lanes16-31 hold K=8..15 / 24..31

  // issue one tile's async copies (6 x b128 per thread, ASYNCcnt += 6)
  auto issue_tile = [&](char* A, char* B, int k) {
    async_copy_b128(A + ldOff,                  gA + k);
    async_copy_b128(A + ldOff + 64 * ROW_BYTES, gA + (size_t)64 * DIN + k);
#pragma unroll
    for (int j = 0; j < 4; ++j)
      async_copy_b128(B + ldOff + j * 64 * ROW_BYTES, gB + (size_t)(j * 64) * DIN + k);
  };

  issue_tile(ldsA0, ldsB0, 0);                      // prologue: fill buffer 0

  for (int it = 0; it < NIT; ++it) {
    const int cur = it & 1;
    char* A = cur ? ldsA1 : ldsA0;
    char* B = cur ? ldsB1 : ldsB0;

    if (it + 1 < NIT) {                             // prefetch next tile into other buffer
      issue_tile(cur ? ldsA0 : ldsA1, cur ? ldsB0 : ldsB1, (it + 1) * BK);
      WAIT_ASYNC(6);                                // current tile's 6 copies complete
    } else {
      WAIT_ASYNC(0);
    }
    __syncthreads();                                // all waves' copies of cur visible

    v16bf af[4], bfr[4];
#pragma unroll
    for (int mt = 0; mt < 4; ++mt) {
      const char* p = A + (wm * 64 + mt * 16 + frow) * ROW_BYTES;
      af[mt] = load_frag(p + aKoff, p + aKoff + 32);
    }
#pragma unroll
    for (int nt = 0; nt < 4; ++nt) {
      const char* p = B + (wn * 64 + nt * 16 + frow) * ROW_BYTES;
      bfr[nt] = load_frag(p + bKoff, p + bKoff + 32);
    }
#pragma unroll
    for (int mt = 0; mt < 4; ++mt)
#pragma unroll
      for (int nt = 0; nt < 4; ++nt)
        acc[mt][nt] = __builtin_amdgcn_wmma_f32_16x16x32_bf16(
            false, af[mt], false, bfr[nt], (short)0, acc[mt][nt], false, false);

    __syncthreads();                                // reads of cur done before it is overwritten
  }

  // epilogue: D layout — VGPR i: lanes0-15 -> M=i, lanes16-31 -> M=i+8; N = lane&15
  const int col  = lane & 15;
  const int rsel = lane >> 4;
#pragma unroll
  for (int nt = 0; nt < 4; ++nt) {
    const int gn = n0 + wn * 64 + nt * 16 + col;
    const float s = scale[gn];
    const float b = bias[gn];
#pragma unroll
    for (int mt = 0; mt < 4; ++mt) {
      const int gm = m0 + wm * 64 + mt * 16 + rsel * 8;
#pragma unroll
      for (int i = 0; i < 8; ++i)
        Out[(size_t)(gm + i) * DOUT + gn] = acc[mt][nt][i] * s + b;
    }
  }
}

// ---------- launcher ----------
extern "C" void kernel_launch(void* const* d_in, const int* in_sizes, int n_in,
                              void* d_out, int out_size, void* d_ws, size_t ws_size,
                              hipStream_t stream) {
  (void)in_sizes; (void)n_in; (void)out_size; (void)ws_size;
  const float* x      = (const float*)d_in[0];   // [8192, 4096]
  const float* weight = (const float*)d_in[1];   // [16384, 4096]
  const float* bias   = (const float*)d_in[2];   // [16384]
  float*       out    = (float*)d_out;           // [8192, 16384]

  char* ws = (char*)d_ws;
  unsigned short* Xb    = (unsigned short*)ws;                                   // 67,108,864 B
  unsigned short* Wb    = (unsigned short*)(ws + (size_t)TOKENS * DIN * 2);      // 134,217,728 B
  float*          scale = (float*)(ws + (size_t)TOKENS * DIN * 2
                                      + (size_t)DOUT * DIN * 2);                 // 65,536 B

  quant_x_kernel<<<(TOKENS * DIN) / (256 * 8), 256, 0, stream>>>(x, Xb);
  quant_w_kernel<<<DOUT, 256, 0, stream>>>(weight, Wb, scale);

  dim3 grid(DOUT / BN, TOKENS / BM);   // 64 x 64 blocks
  bitnet_wmma_kernel<<<grid, 256, SMEM_BYTES, stream>>>(Xb, Wb, scale, bias, out);
}
